// Attention_2190433321184
// MI455X (gfx1250) — matrix operand order repaired and checked by reading, not compiled, not run on hardware
//
#include <hip/hip_runtime.h>

#define B_ 4
#define S_ 2048
#define D_ 1024
#define H_ 16
#define HD_ 64

typedef _Float16       v16h  __attribute__((ext_vector_type(16)));
typedef float          v8f   __attribute__((ext_vector_type(8)));
typedef unsigned short us8   __attribute__((ext_vector_type(8)));
typedef unsigned short us16  __attribute__((ext_vector_type(16)));

__device__ __forceinline__ v8f wmma16(v16h a, v16h b, v8f c) {
  // D = A(16x32 f16) * B(32x16 f16) + C(16x16 f32)
  return __builtin_amdgcn_wmma_f32_16x16x32_f16(
      /*neg_a=*/false, a, /*neg_b=*/false, b,
      /*c_mod=*/(short)0, c, /*reuse_a=*/false, /*reuse_b=*/false);
}

__device__ __forceinline__ unsigned short f2h(float x) {
  return __builtin_bit_cast(unsigned short, (_Float16)x);
}

__device__ __forceinline__ v16h combine16(us8 a, us8 b) {
  us16 u = __builtin_shufflevector(a, b, 0,1,2,3,4,5,6,7,8,9,10,11,12,13,14,15);
  return __builtin_bit_cast(v16h, u);
}

// Low 32 bits of a generic pointer into __shared__ are the LDS byte offset
// (ISA 10.2 aperture rules).
__device__ __forceinline__ unsigned ldsAddr(const void* p) {
  return (unsigned)(size_t)p;
}

// GVS-mode async copy: 16 bytes per lane, global -> LDS, no VGPR landing zone.
// Tracked by ASYNCcnt (invisible to the compiler -> we wait explicitly).
__device__ __forceinline__ void asyncLoadB128(unsigned ldsDst, unsigned gOff,
                                              const void* gBase) {
  asm volatile("global_load_async_to_lds_b128 %0, %1, %2"
               :
               : "v"(ldsDst), "v"(gOff),
                 "s"((unsigned long long)(size_t)gBase)
               : "memory");
}

// -----------------------------------------------------------------------------
// Kernel 1: fused per-head q/k/v projections via WMMA.
//   q[b,h,s,:] = (X_q[b,s,h*64:+64] @ Wq^T + bq) * 1/8   (scale folded in)
//   k[b,h,s,:] =  X_k @ Wk^T + bk
//   v stored TRANSPOSED as vt[b,h,d,s] so the PV B-fragment is a contiguous read.
// One wave = 16 rows of one (b,h). 8 waves per block -> 128 rows.
// -----------------------------------------------------------------------------
__global__ __launch_bounds__(256) void proj_qkv(
    const float* __restrict__ xq, const float* __restrict__ xk,
    const float* __restrict__ xv,
    const float* __restrict__ Wq, const float* __restrict__ bq,
    const float* __restrict__ Wk, const float* __restrict__ bk,
    const float* __restrict__ Wv, const float* __restrict__ bv,
    unsigned short* __restrict__ qh, unsigned short* __restrict__ kh,
    unsigned short* __restrict__ vt)
{
  const int blk  = blockIdx.x;
  const int t    = blk & 15;           // 16 row-tiles of 128 per (b,h)
  const int bh   = blk >> 4;
  const int h    = bh % H_;
  const int b    = bh / H_;
  const int w    = threadIdx.x >> 5;
  const int lane = threadIdx.x & 31;
  const int m    = lane & 15;
  const bool lo  = lane < 16;
  const int s0   = t * 128 + w * 16;

  const float* xs[3] = { xq, xk, xv };
  const float* Ws[3] = { Wq, Wk, Wv };
  const float* bs[3] = { bq, bk, bv };

  #pragma unroll
  for (int mat = 0; mat < 3; ++mat) {
    // A-fragment: row M = lane%16; lo lanes hold K 0..7 & 16..23, hi lanes 8..15 & 24..31
    const float* xrow = xs[mat] + ((size_t)b * S_ + (size_t)(s0 + m)) * D_ + h * HD_;
    v16h aX[2];
    #pragma unroll
    for (int kf = 0; kf < 2; ++kf) {
      const int o = kf * 32 + (lo ? 0 : 8);
      us16 u;
      #pragma unroll
      for (int j = 0; j < 8; ++j) u[j]     = f2h(xrow[o + j]);
      #pragma unroll
      for (int j = 0; j < 8; ++j) u[8 + j] = f2h(xrow[o + 16 + j]);
      aX[kf] = __builtin_bit_cast(v16h, u);
    }
    #pragma unroll
    for (int nb = 0; nb < 4; ++nb) {
      const int n = nb * 16 + m;           // output column (per-lane)
      v8f acc = {0.f,0.f,0.f,0.f,0.f,0.f,0.f,0.f};
      #pragma unroll
      for (int kf = 0; kf < 2; ++kf) {
        // B-fragment: col N = lane%16; lo lanes hold K 0..7 & 16..23, hi lanes 8..15 & 24..31.
        // B[kd][n] = W[n][d] (W^T), so read row n of W.
        const float* wrow = Ws[mat] + (size_t)n * HD_ + kf * 32 + (lo ? 0 : 8);
        us16 u;
        #pragma unroll
        for (int j = 0; j < 16; ++j) u[j] = f2h(wrow[j < 8 ? j : j + 8]);
        acc = wmma16(aX[kf], __builtin_bit_cast(v16h, u), acc);
      }
      const float bias  = bs[mat][n];
      const float scale = (mat == 0) ? 0.125f : 1.0f;  // 1/sqrt(64) folded into q
      #pragma unroll
      for (int i = 0; i < 8; ++i) {        // C-layout: VGPR i -> row i (lo) / i+8 (hi)
        const int row = s0 + i + (lo ? 0 : 8);
        const unsigned short hv = f2h((acc[i] + bias) * scale);
        if (mat == 0)      qh[(((size_t)b * H_ + h) * S_ + row) * HD_ + n] = hv;
        else if (mat == 1) kh[(((size_t)b * H_ + h) * S_ + row) * HD_ + n] = hv;
        else               vt[(((size_t)b * H_ + h) * HD_ + n) * S_ + row] = hv;
      }
    }
  }
}

// -----------------------------------------------------------------------------
// Kernel 2: flash attention. One wave = 16 Q rows; block = 128 rows of one (b,h).
// KV tiles of 32 double-buffered in LDS via async-to-LDS copies (ASYNCcnt);
// online softmax; O accumulated with rescaling.
// -----------------------------------------------------------------------------
__global__ __launch_bounds__(256) void flash_attn(
    const unsigned short* __restrict__ qh, const unsigned short* __restrict__ kh,
    const unsigned short* __restrict__ vt, float* __restrict__ out)
{
  __shared__ __align__(16) unsigned short lds_k[2][32 * 64];   // K tiles  [kv][d]
  __shared__ __align__(16) unsigned short lds_v[2][64 * 32];   // V^T tiles [d][kv]
  __shared__ __align__(16) unsigned short lds_p[8][16 * 32];   // per-wave P bounce

  const int blk  = blockIdx.x;
  const int t    = blk & 15;
  const int bh   = blk >> 4;
  const int h    = bh % H_;
  const int b    = bh / H_;
  const int tid  = threadIdx.x;
  const int w    = tid >> 5;
  const int lane = tid & 31;
  const int m    = lane & 15;
  const bool lo  = lane < 16;
  const int qs   = t * 128 + w * 16;

  const size_t bhBase = (size_t)(b * H_ + h) * S_ * HD_;

  // Q A-fragments (kept in registers for the whole sweep)
  const unsigned short* qrow = qh + bhBase + (size_t)(qs + m) * HD_;
  v16h aQ[2];
  #pragma unroll
  for (int kf = 0; kf < 2; ++kf) {
    const int o = kf * 32 + (lo ? 0 : 8);
    aQ[kf] = combine16(*(const us8*)(qrow + o), *(const us8*)(qrow + o + 16));
  }

  const unsigned short* kbase = kh + bhBase;   // [S][64]
  const unsigned short* vbase = vt + bhBase;   // [64][S] (same extent)

  v8f o0 = {0.f,0.f,0.f,0.f,0.f,0.f,0.f,0.f};
  v8f o1 = o0, o2 = o0, o3 = o0;
  float mrow[8], lrow[8];
  #pragma unroll
  for (int i = 0; i < 8; ++i) { mrow[i] = -1e30f; lrow[i] = 0.f; }

  // cooperative tile coords: 16 B per thread, 8 waves cover each 4 KB tile
  const int kr = tid >> 3, kc = (tid & 7) * 8;   // K tile  [32][64]
  const int vr = tid >> 2, vc = (tid & 3) * 8;   // V^T tile [64][32]

  // prologue: tile 0 -> buffer 0
  asyncLoadB128(ldsAddr(&lds_k[0][kr * 64 + kc]),
                (unsigned)(((0 + kr) * HD_ + kc) * 2), kbase);
  asyncLoadB128(ldsAddr(&lds_v[0][vr * 32 + vc]),
                (unsigned)((vr * S_ + 0 + vc) * 2), vbase);

  for (int kv0 = 0, it = 0; kv0 < S_; kv0 += 32, ++it) {
    const int cur = it & 1;
    __syncthreads();   // all waves done consuming buffer cur^1 (previous iter)
    if (kv0 + 32 < S_) {
      const int nxt = kv0 + 32;
      asyncLoadB128(ldsAddr(&lds_k[cur ^ 1][kr * 64 + kc]),
                    (unsigned)(((nxt + kr) * HD_ + kc) * 2), kbase);
      asyncLoadB128(ldsAddr(&lds_v[cur ^ 1][vr * 32 + vc]),
                    (unsigned)((vr * S_ + nxt + vc) * 2), vbase);
      // async ops complete in order: <=2 outstanding => tile kv0 has landed
      asm volatile("s_wait_asynccnt 0x2" ::: "memory");
    } else {
      asm volatile("s_wait_asynccnt 0x0" ::: "memory");
    }
    __syncthreads();   // tile kv0 visible to every wave

    // S = Q @ K^T : two 16x16 score blocks (kv columns 0..15 and 16..31)
    v8f s0 = {0.f,0.f,0.f,0.f,0.f,0.f,0.f,0.f};
    v8f s1 = s0;
    #pragma unroll
    for (int kf = 0; kf < 2; ++kf) {
      const unsigned short* k0 = &lds_k[cur][(size_t)m        * 64 + kf * 32 + (lo ? 0 : 8)];
      const unsigned short* k1 = &lds_k[cur][(size_t)(16 + m) * 64 + kf * 32 + (lo ? 0 : 8)];
      s0 = wmma16(aQ[kf], combine16(*(const us8*)k0, *(const us8*)(k0 + 16)), s0);
      s1 = wmma16(aQ[kf], combine16(*(const us8*)k1, *(const us8*)(k1 + 16)), s1);
    }

    // online softmax (row reductions across the 16-lane halves)
    #pragma unroll
    for (int i = 0; i < 8; ++i) {
      float mx = fmaxf(s0[i], s1[i]);
      #pragma unroll
      for (int d = 1; d < 16; d <<= 1) mx = fmaxf(mx, __shfl_xor(mx, d, 32));
      const float mn    = fmaxf(mrow[i], mx);
      const float alpha = __expf(mrow[i] - mn);
      mrow[i] = mn;
      const float p0 = __expf(s0[i] - mn);
      const float p1 = __expf(s1[i] - mn);
      lrow[i] = lrow[i] * alpha + p0 + p1;     // per-lane partial; reduced at end
      const int prow = i + (lo ? 0 : 8);
      lds_p[w][prow * 32 + m]      = f2h(p0);
      lds_p[w][prow * 32 + 16 + m] = f2h(p1);
      o0[i] *= alpha; o1[i] *= alpha; o2[i] *= alpha; o3[i] *= alpha;
    }

    // Re-lay P (16x32) into WMMA A-format via per-wave LDS bounce
    const unsigned short* pr = &lds_p[w][(size_t)m * 32 + (lo ? 0 : 16)];
    const v16h aP = combine16(*(const us8*)pr, *(const us8*)(pr + 8));

    // O += P @ V : four 16-wide d-blocks, single K=32 WMMA each
    const unsigned short* v0p = &lds_v[cur][(size_t)(m)      * 32 + (lo ? 0 : 16)];
    const unsigned short* v1p = &lds_v[cur][(size_t)(16 + m) * 32 + (lo ? 0 : 16)];
    const unsigned short* v2p = &lds_v[cur][(size_t)(32 + m) * 32 + (lo ? 0 : 16)];
    const unsigned short* v3p = &lds_v[cur][(size_t)(48 + m) * 32 + (lo ? 0 : 16)];
    o0 = wmma16(aP, combine16(*(const us8*)v0p, *(const us8*)(v0p + 8)), o0);
    o1 = wmma16(aP, combine16(*(const us8*)v1p, *(const us8*)(v1p + 8)), o1);
    o2 = wmma16(aP, combine16(*(const us8*)v2p, *(const us8*)(v2p + 8)), o2);
    o3 = wmma16(aP, combine16(*(const us8*)v3p, *(const us8*)(v3p + 8)), o3);
  }

  // normalize and store fp32 output [B,S,D]; output is never re-read -> NT stores
  #pragma unroll
  for (int i = 0; i < 8; ++i) {
    float l = lrow[i];
    #pragma unroll
    for (int d = 1; d < 16; d <<= 1) l += __shfl_xor(l, d, 32);
    const float inv = 1.0f / l;
    const int row = qs + i + (lo ? 0 : 8);
    float* orow = out + ((size_t)b * S_ + row) * D_ + h * HD_;
    __builtin_nontemporal_store(o0[i] * inv, &orow[ 0 + m]);
    __builtin_nontemporal_store(o1[i] * inv, &orow[16 + m]);
    __builtin_nontemporal_store(o2[i] * inv, &orow[32 + m]);
    __builtin_nontemporal_store(o3[i] * inv, &orow[48 + m]);
  }
}

// -----------------------------------------------------------------------------
extern "C" void kernel_launch(void* const* d_in, const int* in_sizes, int n_in,
                              void* d_out, int out_size, void* d_ws, size_t ws_size,
                              hipStream_t stream) {
  (void)in_sizes; (void)n_in; (void)out_size; (void)ws_size;
  const float* xq = (const float*)d_in[0];
  const float* xk = (const float*)d_in[1];
  const float* xv = (const float*)d_in[2];
  const float* Wq = (const float*)d_in[3];
  const float* bq = (const float*)d_in[4];
  const float* Wk = (const float*)d_in[5];
  const float* bk = (const float*)d_in[6];
  const float* Wv = (const float*)d_in[7];
  const float* bv = (const float*)d_in[8];

  const size_t nE = (size_t)B_ * H_ * S_ * HD_;   // 8,388,608 elems (16 MB fp16 each)
  unsigned short* qh = (unsigned short*)d_ws;
  unsigned short* kh = qh + nE;
  unsigned short* vt = kh + nE;

  dim3 grid(B_ * H_ * (S_ / 128));   // 1024 workgroups
  proj_qkv<<<grid, 256, 0, stream>>>(xq, xk, xv, Wq, bq, Wk, bk, Wv, bv, qh, kh, vt);
  flash_attn<<<grid, 256, 0, stream>>>(qh, kh, vt, (float*)d_out);
}
